// RecurrentDecoder_65317862637795
// MI455X (gfx1250) — matrix operand order repaired and checked by reading, not compiled, not run on hardware
//
#include <hip/hip_runtime.h>
#include <hip/hip_bf16.h>

typedef __bf16 bf16;
typedef bf16  v4bf  __attribute__((ext_vector_type(4)));
typedef bf16  v8bf  __attribute__((ext_vector_type(8)));
typedef bf16  v16bf __attribute__((ext_vector_type(16)));
typedef float v4f   __attribute__((ext_vector_type(4)));
typedef float v8f   __attribute__((ext_vector_type(8)));

#define T_STEPS 2048
#define BATCH   16
#define DMODEL  1024
#define DBOT    128
#define DHID    512
#define LN_EPS  1e-5f

#define EPI_BIAS      0
#define EPI_TANH      1
#define EPI_RELU_BF16 2
#define EPI_LN        3

// ---------------------------------------------------------------------------
// Weight transpose + f32->bf16 convert: dst[n*K + k] = (bf16)src[k*N + n]
// ---------------------------------------------------------------------------
__global__ void transpose_bf16_kernel(const float* __restrict__ src,
                                      bf16* __restrict__ dst, int K, int N) {
    int i = blockIdx.x * blockDim.x + threadIdx.x;
    if (i < K * N) {
        int n = i / K, k = i % K;
        dst[(size_t)n * K + k] = (bf16)src[(size_t)k * N + n];
    }
}

// ---------------------------------------------------------------------------
// Elementwise diagonal recurrence: one thread per (b, d) channel.
// mem_all[t*B+b][d] = mem state BEFORE update at step t; final_mem = mem_T.
// ---------------------------------------------------------------------------
__global__ void mem_scan_kernel(const float* __restrict__ mem0,
                                const unsigned char* __restrict__ mask,
                                const float* __restrict__ wbuf,
                                const float* __restrict__ decay_logit,
                                float* __restrict__ mem_all,
                                float* __restrict__ final_mem) {
    int idx = blockIdx.x * blockDim.x + threadIdx.x;
    if (idx >= BATCH * DBOT) return;
    int b = idx >> 7, d = idx & (DBOT - 1);
    float dec = 1.f / (1.f + __expf(-decay_logit[d]));
    float one_m = 1.f - dec;
    float m = mem0[idx];
    for (int t = 0; t < T_STEPS; ++t) {
        int row = t * BATCH + b;
        mem_all[(size_t)row * DBOT + d] = m;
        if (!mask[row]) m = dec * m + one_m * wbuf[(size_t)row * DBOT + d];
    }
    final_mem[idx] = m;
}

// ---------------------------------------------------------------------------
// Tiled WMMA GEMM: C(16 x N) = A(16 x K) * Bt^T + epilogue, one M-tile/block.
// Block = 128 threads = 4 waves, each wave owns N/64 16x16 output tiles.
// A double-buffered in LDS as bf16 (vectorized f32->bf16 staging).
// Bt is bf16 stored N-major: Bt[n*K + k] = W[k][n] (L2-resident).
// ---------------------------------------------------------------------------
template<int K, int N, bool ABF16, int EPI>
__global__ __launch_bounds__(128)
void gemm16_kernel(const void* __restrict__ Aptr,
                   const bf16* __restrict__ Bt,
                   const float* __restrict__ bias,
                   const float* __restrict__ res,
                   const float* __restrict__ ln_s,
                   const float* __restrict__ ln_b,
                   float* __restrict__ outF,
                   bf16* __restrict__ outB) {
    extern __shared__ char smem[];
    bf16*  sA0    = (bf16*)smem;                              // 16 x 32 bf16
    bf16*  sA1    = (bf16*)(smem + 1024);                     // double buffer
    float* sC     = (float*)(smem + 2048);                    // 16 x N f32
    float* sPart  = (float*)(smem + 2048 + (size_t)16 * N * 4); // 16 x 8
    float* sPart2 = sPart + 128;
    float* sMu    = sPart2 + 128;
    float* sRs    = sMu + 16;

    const int tid  = threadIdx.x;
    const int lane = tid & 31;
    const int wv   = tid >> 5;        // wave 0..3
    const int hl   = lane >> 4;       // half-wave 0/1
    const int l15  = lane & 15;
    const int m0   = blockIdx.x * 16;

    constexpr int TPW    = N / 64;    // 16x16 tiles per wave
    constexpr int KSTEPS = K / 32;

    v8f acc[TPW];
    #pragma unroll
    for (int j = 0; j < TPW; ++j)
        #pragma unroll
        for (int r = 0; r < 8; ++r) acc[j][r] = 0.f;

    const float* Af = (const float*)Aptr;
    const bf16*  Ab = (const bf16*)Aptr;

    // vectorized staging: 128 threads x 4 elems == 16x32 tile, one shot
    const int st_row = tid >> 3;
    const int st_col = (tid & 7) * 4;
    auto stage = [&](int k0, bf16* dst) {
        v4bf v;
        if (ABF16) {
            v = *(const v4bf*)(Ab + (size_t)(m0 + st_row) * K + k0 + st_col);
        } else {
            v4f f = *(const v4f*)(Af + (size_t)(m0 + st_row) * K + k0 + st_col);
            #pragma unroll
            for (int j = 0; j < 4; ++j) v[j] = (bf16)f[j];
        }
        *(v4bf*)(dst + st_row * 32 + st_col) = v;
    };

    stage(0, sA0);
    __syncthreads();

    for (int s = 0; s < KSTEPS; ++s) {
        const int k0 = s * 32;
        bf16* cur = (s & 1) ? sA1 : sA0;
        bf16* nxt = (s & 1) ? sA0 : sA1;
        if (s + 1 < KSTEPS) stage(k0 + 32, nxt);  // prefetch into other buffer

        // A fragment (16-bit 16x32): lanes<16 hold K 0-15,
        // lanes>=16 hold K 16-31 (matches B), row M = lane&15.
        const bf16* ap = cur + l15 * 32 + hl * 16;
        v8bf alo = *(const v8bf*)(ap);
        v8bf ahi = *(const v8bf*)(ap + 8);
        v16bf afrag;
        #pragma unroll
        for (int i = 0; i < 8; ++i) { afrag[i] = alo[i]; afrag[8 + i] = ahi[i]; }

        #pragma unroll
        for (int j = 0; j < TPW; ++j) {
            // B fragment: col N = tile*16 + (lane&15); lanes<16 K 0-15,
            // lanes>=16 K 16-31 -> one contiguous 32B load from N-major Bt.
            int n = (wv * TPW + j) * 16 + l15;
            v16bf bfrag = *(const v16bf*)(Bt + (size_t)n * K + k0 + hl * 16);
            acc[j] = __builtin_amdgcn_wmma_f32_16x16x32_bf16(
                         false, afrag, false, bfrag,
                         (short)0, acc[j], false, false);
        }
        __syncthreads();
    }

    // C layout: VGPR r <-> row r + 8*(lane>=16), col = lane&15
    #pragma unroll
    for (int j = 0; j < TPW; ++j) {
        int ncol = (wv * TPW + j) * 16 + l15;
        #pragma unroll
        for (int r = 0; r < 8; ++r)
            sC[(r + hl * 8) * N + ncol] = acc[j][r];
    }
    __syncthreads();

    constexpr int CHUNKS = (16 * N) / 4;   // float4 chunks

    if constexpr (EPI == EPI_BIAS || EPI == EPI_TANH || EPI == EPI_RELU_BF16) {
        for (int q = tid; q < CHUNKS; q += 128) {
            int i = q * 4;
            int r = i / N, c = i % N;
            v4f v  = *(const v4f*)(sC + i);
            v4f bi = *(const v4f*)(bias + c);
            size_t o = (size_t)(m0 + r) * N + c;
            if constexpr (EPI == EPI_TANH) {
                v4f t;
                #pragma unroll
                for (int j = 0; j < 4; ++j) t[j] = tanhf(v[j] + bi[j]);
                *(v4f*)(outF + o) = t;
            } else if constexpr (EPI == EPI_RELU_BF16) {
                v4bf t;
                #pragma unroll
                for (int j = 0; j < 4; ++j) t[j] = (bf16)fmaxf(v[j] + bi[j], 0.f);
                *(v4bf*)(outB + o) = t;
            } else {
                v4f t;
                #pragma unroll
                for (int j = 0; j < 4; ++j) t[j] = v[j] + bi[j];
                *(v4f*)(outF + o) = t;
            }
        }
    } else {
        // residual add, then row LayerNorm over N columns
        for (int q = tid; q < CHUNKS; q += 128) {
            int i = q * 4;
            int r = i / N, c = i % N;
            v4f v  = *(const v4f*)(sC + i);
            v4f bi = *(const v4f*)(bias + c);
            v4f rs = *(const v4f*)(res + (size_t)(m0 + r) * N + c);
            #pragma unroll
            for (int j = 0; j < 4; ++j) v[j] += bi[j] + rs[j];
            *(v4f*)(sC + i) = v;
        }
        __syncthreads();
        {
            int row = tid >> 3, sub = tid & 7;     // 8 threads per row
            const int per = N / 8;                 // multiple of 4
            float s = 0.f, s2 = 0.f;
            for (int c = sub * per; c < (sub + 1) * per; c += 4) {
                v4f v = *(const v4f*)(sC + row * N + c);
                #pragma unroll
                for (int j = 0; j < 4; ++j) { s += v[j]; s2 += v[j] * v[j]; }
            }
            sPart[row * 8 + sub]  = s;
            sPart2[row * 8 + sub] = s2;
        }
        __syncthreads();
        if (tid < 16) {
            float S = 0.f, S2 = 0.f;
            #pragma unroll
            for (int j = 0; j < 8; ++j) { S += sPart[tid * 8 + j]; S2 += sPart2[tid * 8 + j]; }
            float mu  = S / (float)N;
            float var = S2 / (float)N - mu * mu;
            sMu[tid] = mu;
            sRs[tid] = rsqrtf(var + LN_EPS);
        }
        __syncthreads();
        for (int q = tid; q < CHUNKS; q += 128) {
            int i = q * 4;
            int r = i / N, c = i % N;
            v4f v  = *(const v4f*)(sC + i);
            v4f sc = *(const v4f*)(ln_s + c);
            v4f sb = *(const v4f*)(ln_b + c);
            float mu = sMu[r], rsi = sRs[r];
            v4f t;
            #pragma unroll
            for (int j = 0; j < 4; ++j) t[j] = (v[j] - mu) * rsi * sc[j] + sb[j];
            *(v4f*)(outF + (size_t)(m0 + r) * N + c) = t;
        }
    }
}

// ---------------------------------------------------------------------------
extern "C" void kernel_launch(void* const* d_in, const int* in_sizes, int n_in,
                              void* d_out, int out_size, void* d_ws, size_t ws_size,
                              hipStream_t stream) {
    (void)in_sizes; (void)n_in; (void)out_size; (void)ws_size;

    const float* x    = (const float*)d_in[0];
    const unsigned char* mask = (const unsigned char*)d_in[1];
    const float* mem0 = (const float*)d_in[2];
    const float* Wb   = (const float*)d_in[3];
    const float* bb   = (const float*)d_in[4];
    const float* Wu   = (const float*)d_in[5];
    const float* bu   = (const float*)d_in[6];
    const float* Wr   = (const float*)d_in[7];
    const float* br   = (const float*)d_in[8];
    const float* Ww   = (const float*)d_in[9];
    const float* bw   = (const float*)d_in[10];
    const float* dlg  = (const float*)d_in[11];
    const float* f1   = (const float*)d_in[12];
    const float* fb1  = (const float*)d_in[13];
    const float* f2   = (const float*)d_in[14];
    const float* fb2  = (const float*)d_in[15];
    const float* mls  = (const float*)d_in[16];
    const float* mlb  = (const float*)d_in[17];
    const float* fls  = (const float*)d_in[18];
    const float* flb  = (const float*)d_in[19];
    const float* ols  = (const float*)d_in[20];
    const float* olb  = (const float*)d_in[21];

    float* out       = (float*)d_out;                                  // (T,B,DMODEL)
    float* final_mem = out + (size_t)T_STEPS * BATCH * DMODEL;         // (B,DBOT)

    const size_t MiB = 1u << 20;
    char* ws = (char*)d_ws;
    float* h    = (float*)(ws + 0   * MiB);   // 16 MiB (M x 128)
    float* wbuf = (float*)(ws + 16  * MiB);   // 16 MiB
    float* mall = (float*)(ws + 32  * MiB);   // 16 MiB
    float* h1   = (float*)(ws + 48  * MiB);   // 16 MiB
    float* h2   = (float*)(ws + 64  * MiB);   // 16 MiB
    bf16*  a1   = (bf16*) (ws + 80  * MiB);   // 32 MiB (M x 512 bf16)
    bf16*  WbT  = (bf16*) (ws + 112 * MiB);
    bf16*  WuT  = WbT + (size_t)DMODEL * DBOT;
    bf16*  WrT  = WuT + (size_t)DBOT * DMODEL;
    bf16*  WwT  = WrT + (size_t)DBOT * DBOT;
    bf16*  F1T  = WwT + (size_t)DBOT * DBOT;
    bf16*  F2T  = F1T + (size_t)DBOT * DHID;

    // 0) convert weights to bf16, N-major (transposed)
    {
        struct { const float* s; bf16* d; int K, N; } tp[6] = {
            { Wb, WbT, DMODEL, DBOT }, { Wu, WuT, DBOT, DMODEL },
            { Wr, WrT, DBOT,   DBOT }, { Ww, WwT, DBOT, DBOT   },
            { f1, F1T, DBOT,   DHID }, { f2, F2T, DHID, DBOT   },
        };
        for (int i = 0; i < 6; ++i) {
            int n = tp[i].K * tp[i].N;
            transpose_bf16_kernel<<<(n + 255) / 256, 256, 0, stream>>>(
                tp[i].s, tp[i].d, tp[i].K, tp[i].N);
        }
    }

    const int M = T_STEPS * BATCH;
    dim3 grid(M / 16), block(128);
    auto smem = [](int N) {
        return (size_t)2048 + (size_t)16 * N * 4 + 2 * 128 * 4 + 2 * 16 * 4;
    };

    // 1) h = x @ Wb + bb
    gemm16_kernel<DMODEL, DBOT, false, EPI_BIAS>
        <<<grid, block, smem(DBOT), stream>>>(x, WbT, bb, nullptr, nullptr, nullptr, h, nullptr);
    // 2) write = tanh(h @ Ww + bw)
    gemm16_kernel<DBOT, DBOT, false, EPI_TANH>
        <<<grid, block, smem(DBOT), stream>>>(h, WwT, bw, nullptr, nullptr, nullptr, wbuf, nullptr);
    // 3) diagonal mem recurrence -> mem_all, final_mem
    mem_scan_kernel<<<(BATCH * DBOT + 255) / 256, 256, 0, stream>>>(
        mem0, mask, wbuf, dlg, mall, final_mem);
    // 4) h1 = LN(h + mem_all @ Wr + br)
    gemm16_kernel<DBOT, DBOT, false, EPI_LN>
        <<<grid, block, smem(DBOT), stream>>>(mall, WrT, br, h, mls, mlb, h1, nullptr);
    // 5) a1 = relu(h1 @ ff_w1 + ff_b1)  (bf16)
    gemm16_kernel<DBOT, DHID, false, EPI_RELU_BF16>
        <<<grid, block, smem(DHID), stream>>>(h1, F1T, fb1, nullptr, nullptr, nullptr, nullptr, a1);
    // 6) h2 = LN(h1 + a1 @ ff_w2 + ff_b2)
    gemm16_kernel<DHID, DBOT, true, EPI_LN>
        <<<grid, block, smem(DBOT), stream>>>(a1, F2T, fb2, h1, fls, flb, h2, nullptr);
    // 7) out = LN(x + h2 @ Wu + bu)
    gemm16_kernel<DBOT, DMODEL, false, EPI_LN>
        <<<grid, block, smem(DMODEL), stream>>>(h2, WuT, bu, x, ols, olb, out, nullptr);
}
